// MultiHeadAttention_24051816857694
// MI455X (gfx1250) — matrix operand order repaired and checked by reading, not compiled, not run on hardware
//
#include <hip/hip_runtime.h>
#include <stdint.h>

#define EMBED 512
#define NHEAD 8
#define HDIM  64
#define BATCH 2
#define SEQ   4096

typedef __attribute__((ext_vector_type(16))) _Float16 v16h;
typedef __attribute__((ext_vector_type(8)))  _Float16 v8h;
typedef __attribute__((ext_vector_type(4)))  _Float16 v4h;
typedef __attribute__((ext_vector_type(8)))  float    v8f;
typedef __attribute__((ext_vector_type(4)))  float    v4f;

// ---------------------------------------------------------------------------
// WMMA fragment loaders (CDNA5 16x16x32 f16 layouts, ISA 7.12.2)
// A (16x32, MxK): lanes 0-15 -> M=lane, K = {0..7, 16..23}; lanes 16-31 -> K = {8..15, 24..31}
// B (32x16, KxN): lanes 0-15 -> N=lane, K = {0..7, 16..23}; lanes 16-31 -> K = {8..15, 24..31}
// C/D (16x16):    lanes 0-15 -> N=lane, VGPR r -> M=r; lanes 16-31 -> M=8+r
// ---------------------------------------------------------------------------

__device__ inline v16h load_a_f16(const _Float16* __restrict__ p, int ld,
                                  int m0, int k0, int lane) {
  const int lr = lane & 15, hi = lane >> 4;
  const _Float16* row = p + (size_t)(m0 + lr) * ld + k0 + hi * 8;
  v8h lo = *(const v8h*)(row);
  v8h hh = *(const v8h*)(row + 16);
  v16h r;
#pragma unroll
  for (int i = 0; i < 8; ++i) { r[i] = lo[i]; r[8 + i] = hh[i]; }
  return r;
}

// B fragment from a matrix stored [N][K] row-major (B-transposed storage)
__device__ inline v16h load_bt_f16(const _Float16* __restrict__ p, int ld,
                                   int n0, int k0, int lane) {
  const int lr = lane & 15, hi = lane >> 4;
  const _Float16* row = p + (size_t)(n0 + lr) * ld + k0 + hi * 8;
  v8h a = *(const v8h*)(row);
  v8h b = *(const v8h*)(row + 16);
  v16h r;
#pragma unroll
  for (int i = 0; i < 8; ++i) { r[i] = a[i]; r[8 + i] = b[i]; }
  return r;
}

#define WMMA_F16(a, b, c) \
  __builtin_amdgcn_wmma_f32_16x16x32_f16(false, (a), false, (b), (short)0, (c), false, false)

// ---------------------------------------------------------------------------
// Kernel 0: one-shot f32 -> f16 conversion (keeps v_cvt out of all GEMM loops)
// ---------------------------------------------------------------------------
__global__ __launch_bounds__(256) void cvt_f16_kernel(
    const float* __restrict__ src, _Float16* __restrict__ dst, int n4) {
  const int i = blockIdx.x * 256 + threadIdx.x;
  if (i < n4) {
    v4f v = *(const v4f*)(src + (size_t)i * 4);
    v4h h;
#pragma unroll
    for (int e = 0; e < 4; ++e) h[e] = (_Float16)v[e];
    *(v4h*)(dst + (size_t)i * 4) = h;
  }
}

// ---------------------------------------------------------------------------
// Kernel 1: fused QKV projection.  y = x @ W^T  (NT gemm, both K-contiguous).
//   z=0 -> Q [B,H,T,D] f16    z=1 -> K [B,H,T,D] f16    z=2 -> V^T [B,H,D,T] f16
// One wave computes a 64(M) x 64(N) tile; N-tile == one head (D=64).
// ---------------------------------------------------------------------------
__global__ __launch_bounds__(32) void qkv_proj_kernel(
    const _Float16* __restrict__ xh, const _Float16* __restrict__ wh,
    _Float16* __restrict__ qb, _Float16* __restrict__ kb,
    _Float16* __restrict__ vt) {
  const int lane = threadIdx.x & 31;
  const int lr = lane & 15, hi = lane >> 4;
  const int n0 = blockIdx.x * 64;
  const int m0 = blockIdx.y * 64;
  const int z  = blockIdx.z;
  const _Float16* W = wh + (size_t)z * EMBED * EMBED;

  v8f acc[4][4];
#pragma unroll
  for (int i = 0; i < 4; ++i)
#pragma unroll
    for (int j = 0; j < 4; ++j)
#pragma unroll
      for (int e = 0; e < 8; ++e) acc[i][j][e] = 0.f;

  for (int k0 = 0; k0 < EMBED; k0 += 32) {
    if (k0 + 32 < EMBED)
      __builtin_prefetch(xh + (size_t)(m0 + lr) * EMBED + k0 + 32, 0, 0);
    v16h a[4], b[4];
#pragma unroll
    for (int i = 0; i < 4; ++i) a[i] = load_a_f16(xh, EMBED, m0 + 16 * i, k0, lane);
#pragma unroll
    for (int j = 0; j < 4; ++j) b[j] = load_bt_f16(W, EMBED, n0 + 16 * j, k0, lane);
#pragma unroll
    for (int i = 0; i < 4; ++i)
#pragma unroll
      for (int j = 0; j < 4; ++j) acc[i][j] = WMMA_F16(a[i], b[j], acc[i][j]);
  }

  const int bidx = m0 / SEQ;
  const int h    = n0 / HDIM;
  const size_t headbase = (size_t)(bidx * NHEAD + h);

  if (z < 2) {
    _Float16* out = (z == 0) ? qb : kb;   // [B,H,T,D]
#pragma unroll
    for (int i = 0; i < 4; ++i)
#pragma unroll
      for (int j = 0; j < 4; ++j) {
        const int d  = 16 * j + lr;
        const int t0 = (m0 - bidx * SEQ) + 16 * i + hi * 8;
        _Float16* base = out + (headbase * SEQ + t0) * HDIM + d;
#pragma unroll
        for (int r = 0; r < 8; ++r) base[(size_t)r * HDIM] = (_Float16)acc[i][j][r];
      }
  } else {                                // V^T [B,H,D,T] -> contiguous b128 store
#pragma unroll
    for (int i = 0; i < 4; ++i)
#pragma unroll
      for (int j = 0; j < 4; ++j) {
        const int d  = 16 * j + lr;
        const int t0 = (m0 - bidx * SEQ) + 16 * i + hi * 8;
        v8h pk;
#pragma unroll
        for (int r = 0; r < 8; ++r) pk[r] = (_Float16)acc[i][j][r];
        *(v8h*)(vt + (headbase * HDIM + d) * SEQ + t0) = pk;
      }
  }
}

// ---------------------------------------------------------------------------
// Async stage of one 32-key block: K tile (contiguous 4KB of [T,D]) and
// V^T tile (64 rows x 64B) -> LDS.  256 x 16B chunks each, 2 per thread.
// ---------------------------------------------------------------------------
__device__ inline void stage_kv_async(unsigned ldsK, unsigned ldsV,
                                      const _Float16* __restrict__ Kp,
                                      const _Float16* __restrict__ Vp,
                                      int j0, int tid) {
  const uint64_t gK = (uint64_t)(Kp + (size_t)j0 * HDIM);
#pragma unroll
  for (int it = 0; it < 2; ++it) {
    const int c = tid + it * 128;
    unsigned loff = ldsK + c * 16;
    uint64_t ga   = gK + (uint64_t)c * 16;
    asm volatile("global_load_async_to_lds_b128 %0, %1, off"
                 :: "v"(loff), "v"(ga) : "memory");
  }
#pragma unroll
  for (int it = 0; it < 2; ++it) {
    const int c = tid + it * 128;
    const int d = c >> 2, part = c & 3;
    unsigned loff = ldsV + c * 16;
    uint64_t ga   = (uint64_t)(Vp + (size_t)d * SEQ + j0) + (uint64_t)part * 16;
    asm volatile("global_load_async_to_lds_b128 %0, %1, off"
                 :: "v"(loff), "v"(ga) : "memory");
  }
}

// ---------------------------------------------------------------------------
// Kernel 2: causal flash attention.
// 128-thread block (4 waves) owns a 64-query tile.  Per 32-key block the K/V^T
// tiles are staged into DOUBLE-BUFFERED LDS with global_load_async_to_lds_b128;
// stage i+1 is issued before computing on stage i, waiting only to
// asynccnt<=4 (in-order completion), so DMA overlaps WMMA + softmax.
// ---------------------------------------------------------------------------
__global__ __launch_bounds__(128) void flash_attn_kernel(
    const _Float16* __restrict__ qb, const _Float16* __restrict__ kb,
    const _Float16* __restrict__ vt, _Float16* __restrict__ attn) {
  __shared__ __attribute__((aligned(16))) _Float16 sK[2][32 * HDIM];  // [key][d]
  __shared__ __attribute__((aligned(16))) _Float16 sV[2][HDIM * 32];  // [d][key]
  __shared__ __attribute__((aligned(16))) _Float16 sP[4][16 * 40];    // per-wave P transpose

  const int tid  = threadIdx.x;
  const int wave = tid >> 5, lane = tid & 31;
  const int lr = lane & 15, hi = lane >> 4;
  const int mblk = blockIdx.x * 64;
  const int m0   = mblk + wave * 16;
  const int h = blockIdx.y, b = blockIdx.z;
  const size_t bh = (size_t)b * NHEAD + h;
  const _Float16* Qp = qb + bh * SEQ * HDIM;   // [T,D]
  const _Float16* Kp = kb + bh * SEQ * HDIM;   // [T,D]
  const _Float16* Vp = vt + bh * HDIM * SEQ;   // [D,T]

  const v16h aq0 = load_a_f16(Qp, HDIM, m0, 0, lane);
  const v16h aq1 = load_a_f16(Qp, HDIM, m0, 32, lane);

  v8f o[4];
  float mrun[8], lrun[8];
#pragma unroll
  for (int f = 0; f < 4; ++f)
#pragma unroll
    for (int e = 0; e < 8; ++e) o[f][e] = 0.f;
#pragma unroll
  for (int r = 0; r < 8; ++r) { mrun[r] = -3.0e38f; lrun[r] = 0.f; }

  // low 32 bits of a generic shared-aperture address == LDS byte offset
  const unsigned ldsK0 = (unsigned)(uintptr_t)&sK[0][0];
  const unsigned ldsK1 = (unsigned)(uintptr_t)&sK[1][0];
  const unsigned ldsV0 = (unsigned)(uintptr_t)&sV[0][0];
  const unsigned ldsV1 = (unsigned)(uintptr_t)&sV[1][0];

  const int nkb = (mblk + 64) >> 5;   // causal: key blocks covering keys <= mblk+63

  stage_kv_async(ldsK0, ldsV0, Kp, Vp, 0, tid);        // prologue: stage 0

  for (int kbI = 0; kbI < nkb; ++kbI) {
    const int j0  = kbI * 32;
    const int buf = kbI & 1;

    if (kbI + 1 < nkb) {            // prefetch next block into the other buffer
      stage_kv_async(buf ? ldsK0 : ldsK1, buf ? ldsV0 : ldsV1,
                     Kp, Vp, j0 + 32, tid);
      asm volatile("s_wait_asynccnt 0x4" ::: "memory");  // stage kbI landed
    } else {
      asm volatile("s_wait_asynccnt 0x0" ::: "memory");
    }
    __syncthreads();

    const _Float16* cK = &sK[buf][0];
    const _Float16* cV = &sV[buf][0];

    // ---- S = Q K^T for this wave's 16 queries x 32 keys (B-frags from LDS)
    v8f s[2];
#pragma unroll
    for (int jj = 0; jj < 2; ++jj) {
#pragma unroll
      for (int e = 0; e < 8; ++e) s[jj][e] = 0.f;
      v16h bk0 = load_bt_f16(cK, HDIM, 16 * jj, 0, lane);
      v16h bk1 = load_bt_f16(cK, HDIM, 16 * jj, 32, lane);
      s[jj] = WMMA_F16(aq0, bk0, s[jj]);
      s[jj] = WMMA_F16(aq1, bk1, s[jj]);
    }

    // ---- scale + causal mask + online softmax
#pragma unroll
    for (int r = 0; r < 8; ++r) {
      const int q = m0 + hi * 8 + r;
      float mx = -3.0e38f;
#pragma unroll
      for (int jj = 0; jj < 2; ++jj) {
        const int key = j0 + 16 * jj + lr;
        float v = s[jj][r] * 0.125f;              // 1/sqrt(64)
        v = (key <= q) ? v : -3.0e38f;
        s[jj][r] = v;
        mx = fmaxf(mx, v);
      }
#pragma unroll
      for (int off = 1; off < 16; off <<= 1) mx = fmaxf(mx, __shfl_xor(mx, off, 32));
      const float mnew  = fmaxf(mrun[r], mx);
      const float alpha = __expf(mrun[r] - mnew);
      float rs = 0.f;
#pragma unroll
      for (int jj = 0; jj < 2; ++jj) {
        const float p = __expf(s[jj][r] - mnew);
        s[jj][r] = p;
        rs += p;
      }
#pragma unroll
      for (int off = 1; off < 16; off <<= 1) rs += __shfl_xor(rs, off, 32);
      lrun[r] = lrun[r] * alpha + rs;
      mrun[r] = mnew;
#pragma unroll
      for (int f = 0; f < 4; ++f) o[f][r] *= alpha;
    }

    // ---- P: D-layout -> A-layout transpose through per-wave LDS buffer
    _Float16* pw = &sP[wave][0];
#pragma unroll
    for (int jj = 0; jj < 2; ++jj)
#pragma unroll
      for (int r = 0; r < 8; ++r)
        pw[(hi * 8 + r) * 40 + jj * 16 + lr] = (_Float16)s[jj][r];
    asm volatile("" ::: "memory");   // DS ops are in-order within a wave
    const v16h ap = load_a_f16(pw, 40, 0, 0, lane);

    // ---- O += P V  (V B-frags from LDS)
#pragma unroll
    for (int f = 0; f < 4; ++f) {
      v16h bv = load_bt_f16(cV, 32, f * 16, 0, lane);
      o[f] = WMMA_F16(ap, bv, o[f]);
    }
    __syncthreads();   // all waves done reading buf before it is re-staged
  }

  // ---- normalize and store as [B,T,C] f16 for the output projection
  _Float16* obase = attn + ((size_t)b * SEQ) * EMBED + (size_t)h * HDIM;
#pragma unroll
  for (int f = 0; f < 4; ++f) {
    const int d = f * 16 + lr;
#pragma unroll
    for (int r = 0; r < 8; ++r) {
      const int t = m0 + hi * 8 + r;
      obase[(size_t)t * EMBED + d] = (_Float16)(o[f][r] / lrun[r]);
    }
  }
}

// ---------------------------------------------------------------------------
// Kernel 3: output projection  out = attn @ Wo^T  (all-f16 operands, f32 out)
// ---------------------------------------------------------------------------
__global__ __launch_bounds__(32) void out_proj_kernel(
    const _Float16* __restrict__ attn, const _Float16* __restrict__ woh,
    float* __restrict__ out) {
  const int lane = threadIdx.x & 31;
  const int lr = lane & 15, hi = lane >> 4;
  const int n0 = blockIdx.x * 64;
  const int m0 = blockIdx.y * 64;

  v8f acc[4][4];
#pragma unroll
  for (int i = 0; i < 4; ++i)
#pragma unroll
    for (int j = 0; j < 4; ++j)
#pragma unroll
      for (int e = 0; e < 8; ++e) acc[i][j][e] = 0.f;

  for (int k0 = 0; k0 < EMBED; k0 += 32) {
    v16h a[4], b[4];
#pragma unroll
    for (int i = 0; i < 4; ++i) a[i] = load_a_f16(attn, EMBED, m0 + 16 * i, k0, lane);
#pragma unroll
    for (int j = 0; j < 4; ++j) b[j] = load_bt_f16(woh, EMBED, n0 + 16 * j, k0, lane);
#pragma unroll
    for (int i = 0; i < 4; ++i)
#pragma unroll
      for (int j = 0; j < 4; ++j) acc[i][j] = WMMA_F16(a[i], b[j], acc[i][j]);
  }

#pragma unroll
  for (int i = 0; i < 4; ++i)
#pragma unroll
    for (int j = 0; j < 4; ++j) {
      const int n  = n0 + 16 * j + lr;
      const int mm = m0 + 16 * i + hi * 8;
#pragma unroll
      for (int r = 0; r < 8; ++r) out[(size_t)(mm + r) * EMBED + n] = acc[i][j][r];
    }
}

// ---------------------------------------------------------------------------
extern "C" void kernel_launch(void* const* d_in, const int* in_sizes, int n_in,
                              void* d_out, int out_size, void* d_ws, size_t ws_size,
                              hipStream_t stream) {
  const float* x  = (const float*)d_in[0];
  const float* Wq = (const float*)d_in[1];
  const float* Wk = (const float*)d_in[2];
  const float* Wv = (const float*)d_in[3];
  const float* Wo = (const float*)d_in[4];
  float* out = (float*)d_out;

  const size_t perMat = (size_t)BATCH * NHEAD * SEQ * HDIM;  // 4M f16 elements
  const size_t wElems = (size_t)EMBED * EMBED;               // 256K per weight
  _Float16* qb   = (_Float16*)d_ws;
  _Float16* kbuf = qb + perMat;
  _Float16* vt   = kbuf + perMat;
  _Float16* attn = vt + perMat;           // [B,T,C] f16
  _Float16* xh   = attn + perMat;         // [B*T,C] f16
  _Float16* wh   = xh + perMat;           // Wq,Wk,Wv f16 back-to-back
  _Float16* woh  = wh + 3 * wElems;       // Wo f16        (total ws ~44MB)

  // one-shot f32 -> f16 conversions
  cvt_f16_kernel<<<(int)(perMat / 4 / 256), 256, 0, stream>>>(x, xh, (int)(perMat / 4));
  cvt_f16_kernel<<<(int)(wElems / 4 / 256), 256, 0, stream>>>(Wq, wh, (int)(wElems / 4));
  cvt_f16_kernel<<<(int)(wElems / 4 / 256), 256, 0, stream>>>(Wk, wh + wElems, (int)(wElems / 4));
  cvt_f16_kernel<<<(int)(wElems / 4 / 256), 256, 0, stream>>>(Wv, wh + 2 * wElems, (int)(wElems / 4));
  cvt_f16_kernel<<<(int)(wElems / 4 / 256), 256, 0, stream>>>(Wo, woh, (int)(wElems / 4));

  dim3 g1(EMBED / 64, (BATCH * SEQ) / 64, 3);
  qkv_proj_kernel<<<g1, 32, 0, stream>>>(xh, wh, qb, kbuf, vt);

  dim3 g2(SEQ / 64, NHEAD, BATCH);
  flash_attn_kernel<<<g2, 128, 0, stream>>>(qb, kbuf, vt, attn);

  dim3 g3(EMBED / 64, (BATCH * SEQ) / 64, 1);
  out_proj_kernel<<<g3, 32, 0, stream>>>(attn, woh, out);
}
